// MSTHN2_21199958573612
// MI455X (gfx1250) — matrix operand order repaired and checked by reading, not compiled, not run on hardware
//
#include <hip/hip_runtime.h>
#include <hip/hip_bf16.h>
#include <math.h>

typedef __attribute__((ext_vector_type(16))) _Float16 v16h;
typedef __attribute__((ext_vector_type(8)))  float    v8f;
typedef __attribute__((ext_vector_type(8)))  _Float16 h8;
typedef __attribute__((ext_vector_type(4)))  unsigned u32x4;
typedef __attribute__((ext_vector_type(8)))  int      i32x8;
typedef __attribute__((ext_vector_type(4)))  int      i32x4;

#define L_SEQ   2048
#define D_MODEL 512
#define BATCH   8
#define TOPK    7
#define BM 128
#define BN 128
#define BK 32
#define LDA_PAD 40   // LDS row stride in halves: 32 data + 8 pad (64B + 16B)

// ---------------------------------------------------------------------------
// Fragment loaders per CDNA5 ISA 7.12.2 (wave32, f16 16x16x32).
// A: lane l -> row m=l&15; halves e: k = 8*(l>>4) + (e&7) + (e>=8?16:0)
// B: LDS holds Bs[n][k]; lane l -> col n=l&15; halves e: k = 8*(l>>4) + (e&7) + (e>=8?16:0)
// C: lane l, vgpr r -> row r + 8*(l>>4), col l&15
// ---------------------------------------------------------------------------
__device__ __forceinline__ v16h load_a_frag(const _Float16* As, int row, int lane) {
    const _Float16* p = As + row * LDA_PAD + ((lane >> 4) << 3);
    v16h a;
#pragma unroll
    for (int e = 0; e < 8; ++e) { a[e] = p[e]; a[e + 8] = p[e + 16]; }
    return a;
}
__device__ __forceinline__ v16h load_b_frag(const _Float16* Bs, int col, int lane) {
    const _Float16* p = Bs + col * LDA_PAD + ((lane >> 4) << 3);
    v16h b;
#pragma unroll
    for (int e = 0; e < 8; ++e) { b[e] = p[e]; b[e + 8] = p[e + 16]; }
    return b;
}
__device__ __forceinline__ h8 cvt8(const float* p) {
    float4 a = ((const float4*)p)[0], b = ((const float4*)p)[1];
    h8 v;
    v[0] = (_Float16)a.x; v[1] = (_Float16)a.y; v[2] = (_Float16)a.z; v[3] = (_Float16)a.w;
    v[4] = (_Float16)b.x; v[5] = (_Float16)b.y; v[6] = (_Float16)b.z; v[7] = (_Float16)b.w;
    return v;
}

// ---------------------------------------------------------------------------
// TDM: 2D f16 tile (tile_d0 x tile_d1 elements) global -> LDS, with 16B pad
// per 64B row so LDS row stride = LDA_PAD halves. D# packing per ISA sec 8.
// clang-23 toolchain: 6-arg builtin (g0, g1, g2, g3, extra i32x8, cpol).
// ---------------------------------------------------------------------------
__device__ __forceinline__ unsigned lds_off(const void* p) {
    // generic pointers to LDS carry the LDS byte offset in the low 32 bits
    return (unsigned)(unsigned long long)p;
}
__device__ __forceinline__ void tdm_load_tile_f16(unsigned lds_byte_off,
                                                  const _Float16* tile_start,
                                                  unsigned tile_d0, unsigned tile_d1,
                                                  unsigned tensor_d0, unsigned tensor_d1,
                                                  unsigned stride_elems) {
    unsigned long long ga = (unsigned long long)tile_start;
    u32x4 g0;
    g0[0] = 1u;                                           // count=1, user desc
    g0[1] = lds_byte_off;                                 // lds_addr
    g0[2] = (unsigned)(ga & 0xffffffffu);                 // global_addr[31:0]
    g0[3] = (unsigned)((ga >> 32) & 0x01ffffffu) | (2u << 30);  // addr[56:32], type=2
    i32x8 g1;
    g1[0] = (1 << 16)        // data_size = 2 bytes
          | (1 << 20)        // pad_enable
          | (3 << 22)        // pad_interval: 16 DWORDs (64B)
          | (3 << 25);       // pad_amount:   4 DWORDs (16B)
    g1[1] = (int)((tensor_d0 & 0xffffu) << 16);           // tensor_dim0[15:0]
    g1[2] = (int)((tensor_d0 >> 16) & 0xffffu) | (int)((tensor_d1 & 0xffffu) << 16);
    g1[3] = (int)((tensor_d1 >> 16) & 0xffffu) | (int)(tile_d0 << 16);  // tile_dim0
    g1[4] = (int)tile_d1;                                 // tile_dim1 (tile_dim2=0)
    g1[5] = (int)stride_elems;                            // tensor_dim0_stride lo
    g1[6] = 0;                                            // stride hi / dim1_stride
    g1[7] = 0;
    i32x4 z4 = {0, 0, 0, 0};
    i32x8 z8 = {0, 0, 0, 0, 0, 0, 0, 0};
    __builtin_amdgcn_tensor_load_to_lds(g0, g1, z4, z4, z8, 0);
}

// ---------------------------------------------------------------------------
// C[M,N] = A[M,512] @ W[512,N] + bias  (A,W f32; f16 WMMA, f32 acc)
// 128x128 block, 8 waves (2 row x 4 col), wave tile 64x32 => 8 WMMA / k-step
// ---------------------------------------------------------------------------
template <typename TOUT>
__global__ __launch_bounds__(256)
void gemm_f16wmma(const float* __restrict__ A, const float* __restrict__ W,
                  const float* __restrict__ bias, TOUT* __restrict__ C, int N) {
    __shared__ _Float16 As[BM][LDA_PAD];
    __shared__ _Float16 Bs[BN][LDA_PAD];
    const int tid  = threadIdx.x;
    const int lane = tid & 31;
    const int wave = tid >> 5;
    const int wr = wave >> 2;        // 0..1  (64-row strips)
    const int wc = wave & 3;         // 0..3  (32-col strips)
    const int m_blk = blockIdx.y * BM;
    const int n_blk = blockIdx.x * BN;

    v8f acc[4][2] = {};
    const int ms = tid >> 1, khs = (tid & 1) << 4;   // A staging: 16 halves/thread

    for (int k0 = 0; k0 < D_MODEL; k0 += BK) {
        const float* srcA = A + (size_t)(m_blk + ms) * D_MODEL + k0 + khs;
        if (k0 + BK < D_MODEL) __builtin_prefetch(srcA + BK, 0, 1);
        *(h8*)&As[ms][khs]     = cvt8(srcA);
        *(h8*)&As[ms][khs + 8] = cvt8(srcA + 8);
        // B transposed: Bs[n][k] = W[k0+k][n_blk+n]; n fastest -> coalesced reads
#pragma unroll
        for (int i = 0; i < 16; ++i) {
            int idx = tid + i * 256;
            int kk = idx >> 7, n = idx & 127;
            Bs[n][kk] = (_Float16)W[(size_t)(k0 + kk) * N + n_blk + n];
        }
        __syncthreads();

        v16h af[4], bf[2];
#pragma unroll
        for (int t = 0; t < 4; ++t)
            af[t] = load_a_frag(&As[0][0], wr * 64 + t * 16 + (lane & 15), lane);
#pragma unroll
        for (int t = 0; t < 2; ++t)
            bf[t] = load_b_frag(&Bs[0][0], wc * 32 + t * 16 + (lane & 15), lane);
#pragma unroll
        for (int tr = 0; tr < 4; ++tr)
#pragma unroll
            for (int tc = 0; tc < 2; ++tc)
                acc[tr][tc] = __builtin_amdgcn_wmma_f32_16x16x32_f16(
                    false, af[tr], false, bf[tc], (short)0, acc[tr][tc], false, false);
        __syncthreads();
    }

#pragma unroll
    for (int tr = 0; tr < 4; ++tr) {
#pragma unroll
        for (int tc = 0; tc < 2; ++tc) {
            int col   = n_blk + wc * 32 + tc * 16 + (lane & 15);
            int rbase = m_blk + wr * 64 + tr * 16 + ((lane >> 4) << 3);
            float bv = bias[col];
#pragma unroll
            for (int r = 0; r < 8; ++r)
                C[(size_t)(rbase + r) * N + col] = (TOUT)(acc[tr][tc][r] + bv);
        }
    }
}

// ---------------------------------------------------------------------------
// mean_raw[b,tau] += sum_t q_t . k_{(t-tau)%L}; Q,K are f16 in global.
// Both LDS tiles are straight 2D tile copies -> loaded by the TDM.
// ---------------------------------------------------------------------------
__global__ __launch_bounds__(256)
void gram_mean(const _Float16* __restrict__ Q, const _Float16* __restrict__ K,
               float* __restrict__ mean_raw) {
    __shared__ _Float16 As[BM][LDA_PAD];
    __shared__ _Float16 Bs[BN][LDA_PAD];
    __shared__ float meanLds[L_SEQ];
    const int tid  = threadIdx.x;
    const int lane = tid & 31;
    const int wave = tid >> 5;
    const int wr = wave >> 2;
    const int wc = wave & 3;
    const int b  = blockIdx.y;
    const int t0 = blockIdx.x * BM;
    const _Float16* Qb = Q + (size_t)b * L_SEQ * D_MODEL;
    const _Float16* Kb = K + (size_t)b * L_SEQ * D_MODEL;

    for (int i = tid; i < L_SEQ; i += 256) meanLds[i] = 0.0f;
    __syncthreads();

    for (int s0 = 0; s0 < L_SEQ; s0 += BN) {
        v8f acc[4][2] = {};
        for (int k0 = 0; k0 < D_MODEL; k0 += BK) {
            if (wave == 0) {   // one wave drives the Tensor Data Mover
                tdm_load_tile_f16(lds_off(&As[0][0]),
                                  Qb + (size_t)t0 * D_MODEL + k0,
                                  BK, BM, D_MODEL, L_SEQ, D_MODEL);
                tdm_load_tile_f16(lds_off(&Bs[0][0]),
                                  Kb + (size_t)s0 * D_MODEL + k0,
                                  BK, BN, D_MODEL, L_SEQ, D_MODEL);
                __builtin_amdgcn_s_wait_tensorcnt(0);
            }
            __syncthreads();

            v16h af[4], bf[2];
#pragma unroll
            for (int t = 0; t < 4; ++t)
                af[t] = load_a_frag(&As[0][0], wr * 64 + t * 16 + (lane & 15), lane);
#pragma unroll
            for (int t = 0; t < 2; ++t)
                bf[t] = load_b_frag(&Bs[0][0], wc * 32 + t * 16 + (lane & 15), lane);
#pragma unroll
            for (int tr = 0; tr < 4; ++tr)
#pragma unroll
                for (int tc = 0; tc < 2; ++tc)
                    acc[tr][tc] = __builtin_amdgcn_wmma_f32_16x16x32_f16(
                        false, af[tr], false, bf[tc], (short)0, acc[tr][tc], false, false);
            __syncthreads();
        }
        // diagonal scatter: tau = (t - s) mod L
#pragma unroll
        for (int tr = 0; tr < 4; ++tr) {
#pragma unroll
            for (int tc = 0; tc < 2; ++tc) {
                int s  = s0 + wc * 32 + tc * 16 + (lane & 15);
                int tb = t0 + wr * 64 + tr * 16 + ((lane >> 4) << 3);
#pragma unroll
                for (int r = 0; r < 8; ++r)
                    atomicAdd(&meanLds[(tb + r - s) & (L_SEQ - 1)], acc[tr][tc][r]);
            }
        }
        __syncthreads();
    }
    for (int i = tid; i < L_SEQ; i += 256)
        atomicAdd(&mean_raw[b * L_SEQ + i], meanLds[i]);
}

// ---------------------------------------------------------------------------
__global__ __launch_bounds__(256)
void topk7(const float* __restrict__ mean_raw, int* __restrict__ index_out) {
    __shared__ float avg[L_SEQ];
    __shared__ float sval[256];
    __shared__ int   sidx[256];
    int tid = threadIdx.x;
    for (int i = tid; i < L_SEQ; i += 256) {
        float s = 0.0f;
        for (int b = 0; b < BATCH; ++b) s += mean_raw[b * L_SEQ + i];
        avg[i] = s;
    }
    __syncthreads();
    for (int k = 0; k < TOPK; ++k) {
        float best = -INFINITY; int bi = 0x7fffffff;
        for (int i = tid; i < L_SEQ; i += 256) {
            float v = avg[i];
            if (v > best || (v == best && i < bi)) { best = v; bi = i; }
        }
        sval[tid] = best; sidx[tid] = bi;
        __syncthreads();
        for (int off = 128; off > 0; off >>= 1) {
            if (tid < off) {
                bool take = sval[tid + off] > sval[tid] ||
                            (sval[tid + off] == sval[tid] && sidx[tid + off] < sidx[tid]);
                if (take) { sval[tid] = sval[tid + off]; sidx[tid] = sidx[tid + off]; }
            }
            __syncthreads();
        }
        if (tid == 0) { index_out[k] = sidx[0]; avg[sidx[0]] = -INFINITY; }
        __syncthreads();
    }
}

__global__ void softmax_w(const float* __restrict__ mean_raw,
                          const int* __restrict__ index,
                          float* __restrict__ tc) {
    int b = threadIdx.x;
    if (b < BATCH) {
        float w[TOPK], mx = -INFINITY;
        for (int k = 0; k < TOPK; ++k) {
            w[k] = mean_raw[b * L_SEQ + index[k]] * (1.0f / (float)D_MODEL);
            mx = fmaxf(mx, w[k]);
        }
        float s = 0.0f;
        for (int k = 0; k < TOPK; ++k) { w[k] = expf(w[k] - mx); s += w[k]; }
        for (int k = 0; k < TOPK; ++k) tc[b * TOPK + k] = w[k] / s;
    }
}

__global__ __launch_bounds__(256)
void gather_agg(const float* __restrict__ Vp, const float* __restrict__ tc,
                const int* __restrict__ index, float* __restrict__ outp) {
    __shared__ float stc[BATCH * TOPK];
    __shared__ int   sdx[TOPK];
    if (threadIdx.x < BATCH * TOPK) stc[threadIdx.x] = tc[threadIdx.x];
    if (threadIdx.x < TOPK)         sdx[threadIdx.x] = index[threadIdx.x];
    __syncthreads();
    size_t i = (size_t)blockIdx.x * blockDim.x + threadIdx.x;
    if (i < (size_t)BATCH * L_SEQ * D_MODEL) {
        int d = (int)(i & (D_MODEL - 1));
        int l = (int)((i >> 9) & (L_SEQ - 1));
        int b = (int)(i >> 20);
        float a = 0.0f;
#pragma unroll
        for (int k = 0; k < TOPK; ++k) {
            int src = (l + sdx[k]) & (L_SEQ - 1);
            a += stc[b * TOPK + k] * Vp[((size_t)b << 20) + ((size_t)src << 9) + d];
        }
        outp[i] = a;
    }
}

// ---------------------------------------------------------------------------
extern "C" void kernel_launch(void* const* d_in, const int* in_sizes, int n_in,
                              void* d_out, int out_size, void* d_ws, size_t ws_size,
                              hipStream_t stream) {
    const float* queries = (const float*)d_in[0];
    const float* keys    = (const float*)d_in[1];
    const float* values  = (const float*)d_in[2];
    const float* Wq = (const float*)d_in[3];  const float* bq = (const float*)d_in[4];
    const float* Wk = (const float*)d_in[5];  const float* bk = (const float*)d_in[6];
    const float* Wv = (const float*)d_in[7];  const float* bv = (const float*)d_in[8];
    const float* Wo = (const float*)d_in[9];  const float* bo = (const float*)d_in[10];

    const size_t NEL  = (size_t)BATCH * L_SEQ * D_MODEL;
    const size_t HMAT = NEL * sizeof(_Float16);   // 16 MB
    const size_t FMAT = NEL * sizeof(float);      // 32 MB
    char* ws = (char*)d_ws;
    _Float16* Qh      = (_Float16*)(ws);
    _Float16* Kh      = (_Float16*)(ws + HMAT);
    float*    Vp      = (float*)   (ws + 2 * HMAT);
    float*    outp    = (float*)   (ws + 2 * HMAT + FMAT);
    float*    mean_raw= (float*)   (ws + 2 * HMAT + 2 * FMAT);
    int*      index   = (int*)     (ws + 2 * HMAT + 2 * FMAT + (size_t)BATCH * L_SEQ * 4);
    float*    tc      = (float*)   (ws + 2 * HMAT + 2 * FMAT + (size_t)BATCH * L_SEQ * 4 + 64);

    const int M = BATCH * L_SEQ;                 // 16384
    dim3 ggrid(D_MODEL / BN, M / BM);            // (4, 128)

    gemm_f16wmma<_Float16><<<ggrid, 256, 0, stream>>>(queries, Wq, bq, Qh, D_MODEL);
    gemm_f16wmma<_Float16><<<ggrid, 256, 0, stream>>>(keys,    Wk, bk, Kh, D_MODEL);
    gemm_f16wmma<float>   <<<ggrid, 256, 0, stream>>>(values,  Wv, bv, Vp, D_MODEL);

    (void)hipMemsetAsync(mean_raw, 0, (size_t)BATCH * L_SEQ * 4, stream);
    gram_mean<<<dim3(L_SEQ / BM, BATCH), 256, 0, stream>>>(Qh, Kh, mean_raw);

    topk7<<<1, 256, 0, stream>>>(mean_raw, index);
    softmax_w<<<1, 32, 0, stream>>>(mean_raw, index, tc);

    gather_agg<<<(int)((NEL + 255) / 256), 256, 0, stream>>>(Vp, tc, index, outp);

    gemm_f16wmma<float><<<ggrid, 256, 0, stream>>>(outp, Wo, bo, (float*)d_out, D_MODEL);
}
